// FrFTLoss_47863115547227
// MI455X (gfx1250) — matrix operand order repaired and checked by reading, not compiled, not run on hardware
//
#include <hip/hip_runtime.h>
#include <hip/hip_bf16.h>

// ---------------------------------------------------------------------------
// FrFT loss for MI455X (gfx1250, wave32, WMMA).
//   part 1: CE(logits, targets)                     -- bandwidth trivial
//   part 2: for alpha in {0.5,1.0,1.5}: CE(|FrFT_alpha(x)|, targets)
//           == 6 real GEMMs [16384,1000]x[1000,1000] via v_wmma_f32_16x16x32_f16
// Twiddle matrices (12.6 MB f16) are built on device into d_ws and stay L2
// resident (192 MB L2). GEMM + magnitude + row softmax are fused per 16-row
// strip so no [3,B,N] intermediate hits HBM.
//
// Round-1 fixes: column dim padded to 1024 (64 tiles) so every wave owns 8
// full tiles -> no EXEC-masked tail around WMMAs; A chunk double-buffered in
// LDS so the global fetch of chunk k+1 flies under the 16 WMMAs of chunk k;
// all 16 B-fragment loads per k-step share one vaddr + immediate offsets.
// ---------------------------------------------------------------------------

typedef __attribute__((ext_vector_type(16))) _Float16 v16h;
typedef __attribute__((ext_vector_type(8)))  _Float16 v8h;
typedef __attribute__((ext_vector_type(8)))  float    v8f;

#define FR_N     1000          // class dim
#define FR_KPAD  1024          // padded K (inner) dim
#define FR_CPAD  1024          // padded output-column dim (64 tiles of 16)
#define FR_NTILE 64            // column tiles (8 per wave, no tail)
#define PI_D     3.14159265358979323846

// ---------------------------------------------------------------- accumulators
__global__ void frft_zero_acc(float* acc) {
    if (threadIdx.x < 4) acc[threadIdx.x] = 0.0f;
}

// ------------------------------------------------------- build twiddle matrices
// Layout: Kmat[a][plane][kcol][m], plane 0 = cos, 1 = sin, f16, 1/N folded in except for alpha 1.0.
__global__ void frft_build_kernels(_Float16* __restrict__ Kmat) {
    int idx = blockIdx.x * blockDim.x + threadIdx.x;
    const int total = 3 * FR_CPAD * FR_KPAD;
    if (idx >= total) return;
    int m = idx % FR_KPAD;
    int k = (idx / FR_KPAD) % FR_CPAD;
    int a = idx / (FR_KPAD * FR_CPAD);

    _Float16 re = (_Float16)0.0f, im = (_Float16)0.0f;
    if (m < FR_N && k < FR_N) {
        double th, scale;
        if (a == 1) {  // alpha == 1.0 -> plain DFT, no 1/N (matches jnp.fft.fft)
            th    = -2.0 * PI_D * (double)m * (double)k / (double)FR_N;
            scale = 1.0;
        } else {
            double alpha = (a == 0) ? 0.5 : 1.5;
            double inv   = 1.0 / ((double)FR_N * (double)FR_N);
            th = (PI_D * alpha * 0.5) * ((double)m * m + (double)k * k) * inv
               - 2.0 * PI_D * alpha * (double)m * (double)k * inv;
            scale = 1.0 / (double)FR_N;
        }
        re = (_Float16)(cos(th) * scale);
        im = (_Float16)(sin(th) * scale);
    }
    size_t base = (size_t)a * 2 * FR_CPAD * FR_KPAD;
    Kmat[base + (size_t)k * FR_KPAD + m] = re;
    Kmat[base + (size_t)FR_CPAD * FR_KPAD + (size_t)k * FR_KPAD + m] = im;
}

// ----------------------------------------------------------- CE over raw logits
// One wave32 per row, shuffle reductions.
__global__ __launch_bounds__(256)
void frft_ce_logits(const float* __restrict__ x, const int* __restrict__ tgt,
                    float* __restrict__ acc, int Brows) {
    int wave = (int)((blockIdx.x * blockDim.x + threadIdx.x) >> 5);
    int lane = threadIdx.x & 31;
    if (wave >= Brows) return;
    const float* xr = x + (size_t)wave * FR_N;

    float mx = -1e30f;
    for (int c = lane; c < FR_N; c += 32) mx = fmaxf(mx, xr[c]);
    for (int o = 16; o > 0; o >>= 1) mx = fmaxf(mx, __shfl_xor(mx, o));

    float s = 0.0f;
    for (int c = lane; c < FR_N; c += 32) s += __expf(xr[c] - mx);
    for (int o = 16; o > 0; o >>= 1) s += __shfl_xor(s, o);

    if (lane == 0) {
        int tg = tgt[wave];
        atomicAdd(&acc[0], -(xr[tg] - mx - __logf(s)));
    }
}

// --------------------------------------------- fused FrFT GEMM + |.| + CE strip
// grid = (B/16, 3 alphas), block = 256 threads (8 waves).
// Each block: 16 rows of x  X  full 1024 output columns for one alpha.
__global__ __launch_bounds__(256)
void frft_gemm_ce(const float* __restrict__ x, const int* __restrict__ tgt,
                  const _Float16* __restrict__ Kmat, float* __restrict__ acc) {
    // LDS: double-buffered A chunk (stride 40 halves -> conflict-free,
    //      16B-aligned rows), magnitude strip, reduction scratch. ~36 KB.
    __shared__ _Float16 Ash[2][16][40];
    __shared__ _Float16 magsh[16][FR_CPAD];
    __shared__ float    redv[16][16];
    __shared__ float    rowstat[16];

    const int a    = blockIdx.y;
    const int b0   = blockIdx.x * 16;
    const int tid  = threadIdx.x;
    const int lane = tid & 31;
    const int w    = tid >> 5;

    v8f accre[8], accim[8];
    v8f zero = {};
#pragma unroll
    for (int i = 0; i < 8; ++i) { accre[i] = zero; accim[i] = zero; }

    // Fragment addressing (contiguous K per lane half, same order for A and B, wave32):
    const int ar  = lane & 15;            // A row
    const int akb = (lane >> 4) * 16;     // A K sub-base
    const int bn  = lane & 15;            // B column within tile
    const int bkb = (lane >> 4) << 4;     // B K sub-base (contiguous 16 halves)

    // One B base pointer per k-step; tile stride (i*256KiB) and cos->sin plane
    // stride (2MiB) both fit the signed 24-bit instruction offset.
    const _Float16* bbase = Kmat + (size_t)a * 2 * FR_CPAD * FR_KPAD
                                 + (size_t)(w * 16 + bn) * FR_KPAD + bkb;
    const size_t TILE_STRIDE  = (size_t)128 * FR_KPAD;      // 8 waves apart
    const size_t PLANE_STRIDE = (size_t)FR_CPAD * FR_KPAD;  // cos -> sin

    // Cooperative A-chunk fetch: 16x32 halves, 2 floats per thread.
    const int aidx = tid * 2;
    const int arow = aidx >> 5, acol = aidx & 31;
    const float* xrow = x + (size_t)(b0 + arow) * FR_N;

    // preload chunk 0
    {
        float f0 = (acol     < FR_N) ? xrow[acol]     : 0.0f;
        float f1 = (acol + 1 < FR_N) ? xrow[acol + 1] : 0.0f;
        Ash[0][arow][acol]     = (_Float16)f0;
        Ash[0][arow][acol + 1] = (_Float16)f1;
    }
    __syncthreads();

    const int NKC = FR_KPAD / 32;  // 32 k-steps
    for (int kci = 0; kci < NKC; ++kci) {
        const int cur = kci & 1;

        // issue global fetch of chunk kci+1 (lands under the WMMAs below)
        float nf0 = 0.0f, nf1 = 0.0f;
        if (kci + 1 < NKC) {
            int col = (kci + 1) * 32 + acol;
            nf0 = (col     < FR_N) ? xrow[col]     : 0.0f;
            nf1 = (col + 1 < FR_N) ? xrow[col + 1] : 0.0f;
        }

        // A fragment from LDS: per-lane contiguous K block {akb..+15}
        v8h alo = *(const v8h*)&Ash[cur][ar][akb];
        v8h ahi = *(const v8h*)&Ash[cur][ar][akb + 8];
        v16h afrag;
#pragma unroll
        for (int q = 0; q < 8; ++q) { afrag[q] = alo[q]; afrag[q + 8] = ahi[q]; }

        const _Float16* bptr = bbase + (size_t)kci * 32;
#pragma unroll
        for (int i = 0; i < 8; ++i) {
            v16h bre = *(const v16h*)(bptr + (size_t)i * TILE_STRIDE);
            v16h bim = *(const v16h*)(bptr + (size_t)i * TILE_STRIDE + PLANE_STRIDE);
            accre[i] = __builtin_amdgcn_wmma_f32_16x16x32_f16(
                false, afrag, false, bre, (short)0, accre[i], false, false);
            accim[i] = __builtin_amdgcn_wmma_f32_16x16x32_f16(
                false, afrag, false, bim, (short)0, accim[i], false, false);
        }

        // stage next chunk into the other buffer (safe: its readers synced at
        // the end of the previous iteration)
        if (kci + 1 < NKC) {
            Ash[1 - cur][arow][acol]     = (_Float16)nf0;
            Ash[1 - cur][arow][acol + 1] = (_Float16)nf1;
        }
        __syncthreads();
    }

    // epilogue: magnitude into LDS (C/D layout: M = j + 8*(lane/16), N = lane&15)
    {
        int hi = lane >> 4, n = lane & 15;
#pragma unroll
        for (int i = 0; i < 8; ++i) {
            int t = w + 8 * i;
#pragma unroll
            for (int j = 0; j < 8; ++j) {
                float re = accre[i][j], im = accim[i][j];
                magsh[j + 8 * hi][t * 16 + n] = (_Float16)sqrtf(re * re + im * im);
            }
        }
    }
    __syncthreads();

    // fused cross-entropy over the 16-row strip: 16 threads per row
    int r = tid >> 4, j = tid & 15;
    float mx = -1e30f;
    for (int c = j; c < FR_N; c += 16) mx = fmaxf(mx, (float)magsh[r][c]);
    redv[r][j] = mx;
    __syncthreads();
    if (j == 0) {
        float m2 = redv[r][0];
#pragma unroll
        for (int q = 1; q < 16; ++q) m2 = fmaxf(m2, redv[r][q]);
        rowstat[r] = m2;
    }
    __syncthreads();
    float rm = rowstat[r];
    float s = 0.0f;
    for (int c = j; c < FR_N; c += 16) s += __expf((float)magsh[r][c] - rm);
    redv[r][j] = s;
    __syncthreads();
    if (j == 0) {
        float ss = 0.0f;
#pragma unroll
        for (int q = 0; q < 16; ++q) ss += redv[r][q];
        int tg = tgt[b0 + r];
        float y = (float)magsh[r][tg];
        atomicAdd(&acc[1 + a], -(y - rm - __logf(ss)));
    }
}

// --------------------------------------------------------------------- combine
__global__ void frft_finalize(const float* __restrict__ acc,
                              float* __restrict__ out, int Brows) {
    if (threadIdx.x == 0 && blockIdx.x == 0) {
        float invB = 1.0f / (float)Brows;
        float ce   = acc[0] * invB;
        float frft = (acc[1] + acc[2] + acc[3]) * invB * (1.0f / 3.0f);
        out[0] = ce + 0.5f * frft;
    }
}

// ----------------------------------------------------------------------- launch
extern "C" void kernel_launch(void* const* d_in, const int* in_sizes, int n_in,
                              void* d_out, int out_size, void* d_ws, size_t ws_size,
                              hipStream_t stream) {
    const float* logits  = (const float*)d_in[0];
    const int*   targets = (const int*)d_in[1];
    const int Brows = in_sizes[1];               // 16384

    // workspace: twiddle matrices (12.6 MB f16) then 4 f32 loss accumulators
    const size_t kmatElems = (size_t)3 * 2 * FR_CPAD * FR_KPAD;
    _Float16* Kmat = (_Float16*)d_ws;
    float*    acc  = (float*)((char*)d_ws + kmatElems * sizeof(_Float16));
    float*    out  = (float*)d_out;

    frft_zero_acc<<<1, 64, 0, stream>>>(acc);

    const int total = 3 * FR_CPAD * FR_KPAD;
    frft_build_kernels<<<(total + 255) / 256, 256, 0, stream>>>(Kmat);

    frft_ce_logits<<<(Brows + 7) / 8, 256, 0, stream>>>(logits, targets, acc, Brows);

    dim3 grid(Brows / 16, 3);
    frft_gemm_ce<<<grid, 256, 0, stream>>>(logits, targets, Kmat, acc);

    frft_finalize<<<1, 1, 0, stream>>>(acc, out, Brows);
}
